// DeepEncoderBlock_4028679324318
// MI455X (gfx1250) — matrix operand order repaired and checked by reading, not compiled, not run on hardware
//
#include <hip/hip_runtime.h>
#include <stdint.h>

typedef __bf16 bf16_t;
typedef __attribute__((ext_vector_type(16))) __bf16 v16bf;
typedef __attribute__((ext_vector_type(8)))  __bf16 v8bf;
typedef __attribute__((ext_vector_type(8)))  float  v8f;

#define NB 4
#define CH 64
#define NQ 8
#define HH 256
#define WW 256
#define PLANE (HH * WW)

__device__ __forceinline__ float waveSum(float v) {
#pragma unroll
  for (int off = 16; off > 0; off >>= 1) v += __shfl_xor(v, off, 32);
  return v;
}

// CDNA5 async global->LDS copy (ASYNCcnt path), 16 bytes per lane.
// Per CDNA5 ISA: each lane copies 16 bytes from its global address to its LDS address.
// Flat pointers to __shared__ carry the LDS byte offset in their low 32 bits.
__device__ __forceinline__ void async_copy_b128(void* lds, const void* gaddr) {
  asm volatile("global_load_async_to_lds_b128 %0, %1, off"
               :: "v"((uint32_t)(uintptr_t)lds), "v"(gaddr)
               : "memory");
}
__device__ __forceinline__ void async_wait0() {
  asm volatile("s_wait_asynccnt 0x0" ::: "memory");
}

// ---------------------------------------------------------------------------
// 0) zero the stats accumulators (per-(b,c) sum / sumsq), 512 floats
// ---------------------------------------------------------------------------
__global__ void k_zero(float* __restrict__ stats) {
  int i = blockIdx.x * 256 + threadIdx.x;
  if (i < 512) stats[i] = 0.f;
}

// ---------------------------------------------------------------------------
// 1) fused axial depthwise conv + residual + relu + 64x64 pointwise + stats
//    one block = one (b, row), 256 threads = the 256 columns
// ---------------------------------------------------------------------------
__global__ __launch_bounds__(256) void k_conv(
    const float* __restrict__ x,
    const float* __restrict__ hwv, const float* __restrict__ hbv,
    const float* __restrict__ vwv, const float* __restrict__ vbv,
    const float* __restrict__ pw, const float* __restrict__ pwb,
    float* __restrict__ y, float* __restrict__ stats) {
  const int bid = blockIdx.x;
  const int b = bid >> 8, row = bid & 255;
  const int t = threadIdx.x;

  __shared__ float s_pw[CH * CH];
  __shared__ float s_hw[CH * 5], s_vw[CH * 5];
  __shared__ float s_hb[CH], s_vb[CH], s_pwb[CH];
  __shared__ float s_xs[5 * WW];
  __shared__ float s_stats[2 * CH];

  for (int i = t; i < CH * CH; i += 256) s_pw[i] = pw[i];
  for (int i = t; i < CH * 5; i += 256) { s_hw[i] = hwv[i]; s_vw[i] = vwv[i]; }
  if (t < CH) { s_hb[t] = hbv[t]; s_vb[t] = vbv[t]; s_pwb[t] = pwb[t]; }
  if (t < 2 * CH) s_stats[t] = 0.f;

  float acc[CH];
#pragma unroll
  for (int o = 0; o < CH; ++o) acc[o] = 0.f;

  for (int c = 0; c < CH; ++c) {
    __syncthreads();
    const float* xc = x + (size_t)(b * CH + c) * PLANE;
#pragma unroll
    for (int r = 0; r < 5; ++r) {
      int yy = row + r - 2;
      s_xs[r * WW + t] = (yy >= 0 && yy < HH) ? xc[(size_t)yy * WW + t] : 0.f;
    }
    __syncthreads();
    float h = 0.f;
#pragma unroll
    for (int tap = 0; tap < 5; ++tap) {
      int xx = t + tap - 2;
      float xv = (xx >= 0 && xx < WW) ? s_xs[2 * WW + xx] : 0.f;
      h += xv * s_hw[c * 5 + tap];
    }
    float v = 0.f;
#pragma unroll
    for (int tap = 0; tap < 5; ++tap) v += s_xs[tap * WW + t] * s_vw[c * 5 + tap];
    float a = h + s_hb[c] + v + s_vb[c] + s_xs[2 * WW + t];
    a = fmaxf(a, 0.f);
#pragma unroll
    for (int o = 0; o < CH; ++o) acc[o] += s_pw[o * CH + c] * a;
  }

  float* yb = y + (size_t)(b * CH) * PLANE + (size_t)row * WW + t;
#pragma unroll
  for (int o = 0; o < CH; ++o) {
    float val = acc[o] + s_pwb[o];
    yb[(size_t)o * PLANE] = val;
    float s1 = waveSum(val);
    float s2 = waveSum(val * val);
    if ((t & 31) == 0) {
      atomicAdd(&s_stats[2 * o], s1);
      atomicAdd(&s_stats[2 * o + 1], s2);
    }
  }
  __syncthreads();
  if (t < 2 * CH) atomicAdd(&stats[b * 2 * CH + t], s_stats[t]);
}

// ---------------------------------------------------------------------------
// 2) fold hybrid norm (0.7*IN + 0.3*BN) into per-(b,c) affine y*A + B
// ---------------------------------------------------------------------------
__global__ void k_coef(const float* __restrict__ stats,
                       const float* __restrict__ bnw, const float* __restrict__ bnb,
                       float* __restrict__ cA, float* __restrict__ cB) {
  int c = threadIdx.x;
  if (c >= CH) return;
  const float Ni = (float)PLANE, Nbt = (float)(NB * PLANE), eps = 1e-5f;
  float bs = 0.f, bsq = 0.f, im[NB], ir[NB];
  for (int b = 0; b < NB; ++b) {
    float s = stats[b * 2 * CH + 2 * c];
    float sq = stats[b * 2 * CH + 2 * c + 1];
    bs += s; bsq += sq;
    float mu = s / Ni;
    float var = sq / Ni - mu * mu;
    im[b] = mu; ir[b] = rsqrtf(var + eps);
  }
  float bm = bs / Nbt;
  float bv = bsq / Nbt - bm * bm;
  float br = rsqrtf(bv + eps);
  float w = bnw[c], bb = bnb[c];
  for (int b = 0; b < NB; ++b) {
    cA[b * CH + c] = 0.7f * ir[b] + 0.3f * br * w;
    cB[b * CH + c] = -0.7f * im[b] * ir[b] + 0.3f * (bb - bm * br * w);
  }
}

// ---------------------------------------------------------------------------
// 3) normalize in place + q/k/v projections.
//    Emits q,k (f32) and v (bf16) in NCHW, plus column-major transposed
//    copies qT,kT,vT laid out [b][w][c][len] so the height-attention blocks
//    get fully contiguous working sets (async-LDS friendly).
// ---------------------------------------------------------------------------
__global__ __launch_bounds__(256) void k_norm_proj(
    float* __restrict__ y,
    const float* __restrict__ qw, const float* __restrict__ qb,
    const float* __restrict__ kw, const float* __restrict__ kb,
    const float* __restrict__ vw, const float* __restrict__ vb,
    const float* __restrict__ cA, const float* __restrict__ cB,
    float* __restrict__ qout, float* __restrict__ kout,
    unsigned short* __restrict__ vout_,
    float* __restrict__ qT, float* __restrict__ kT,
    unsigned short* __restrict__ vT_) {
  bf16_t* vout = (bf16_t*)vout_;
  bf16_t* vT = (bf16_t*)vT_;
  const int bid = blockIdx.x;
  const int b = bid >> 8, row = bid & 255;
  const int t = threadIdx.x;

  __shared__ float s_qw[NQ * CH], s_kw[NQ * CH], s_vw[CH * CH];
  __shared__ float s_qb[NQ], s_kb[NQ], s_vb[CH], s_cA[CH], s_cB[CH];
  for (int i = t; i < NQ * CH; i += 256) { s_qw[i] = qw[i]; s_kw[i] = kw[i]; }
  for (int i = t; i < CH * CH; i += 256) s_vw[i] = vw[i];
  if (t < NQ) { s_qb[t] = qb[t]; s_kb[t] = kb[t]; }
  if (t < CH) { s_vb[t] = vb[t]; s_cA[t] = cA[b * CH + t]; s_cB[t] = cB[b * CH + t]; }
  __syncthreads();

  float* ybase = y + (size_t)(b * CH) * PLANE + (size_t)row * WW + t;
  float qa[NQ], ka[NQ], va[CH];
#pragma unroll
  for (int o = 0; o < NQ; ++o) { qa[o] = 0.f; ka[o] = 0.f; }
#pragma unroll
  for (int o = 0; o < CH; ++o) va[o] = 0.f;

  for (int c = 0; c < CH; ++c) {
    float yv = ybase[(size_t)c * PLANE];
    float yn = yv * s_cA[c] + s_cB[c];
    ybase[(size_t)c * PLANE] = yn;
#pragma unroll
    for (int o = 0; o < NQ; ++o) {
      qa[o] += s_qw[o * CH + c] * yn;
      ka[o] += s_kw[o * CH + c] * yn;
    }
#pragma unroll
    for (int o = 0; o < CH; ++o) va[o] += s_vw[o * CH + c] * yn;
  }

  float* qbase = qout + (size_t)(b * NQ) * PLANE + (size_t)row * WW + t;
  float* kbase = kout + (size_t)(b * NQ) * PLANE + (size_t)row * WW + t;
  const size_t colQ = ((size_t)(b * WW + t) * NQ) * HH + row;   // qT/kT index base
  const size_t colV = ((size_t)(b * WW + t) * CH) * HH + row;   // vT index base
#pragma unroll
  for (int o = 0; o < NQ; ++o) {
    float qq = qa[o] + s_qb[o];
    float kk = ka[o] + s_kb[o];
    qbase[(size_t)o * PLANE] = qq;
    kbase[(size_t)o * PLANE] = kk;
    qT[colQ + (size_t)o * HH] = qq;
    kT[colQ + (size_t)o * HH] = kk;
  }
  bf16_t* vbase = vout + (size_t)(b * CH) * PLANE + (size_t)row * WW + t;
#pragma unroll
  for (int o = 0; o < CH; ++o) {
    float vv = va[o] + s_vb[o];
    vbase[(size_t)o * PLANE] = (bf16_t)vv;
    vT[colV + (size_t)o * HH] = (bf16_t)vv;
  }
}

// ---------------------------------------------------------------------------
// attention core notes:
// WMMA bf16 fragment packing used here (contiguous k, same order for A and B):
//  A-frag 16x32: lane<16 -> row=lane, halfs = K[j0..j0+15];
//                lane>=16 -> row=lane-16, K offset +16.
//  B-frag 32x16: lane<16 -> col=lane, halfs = K[j0..j0+15]; lane>=16 -> K+16.
// Staging uses global_load_async_to_lds_b128 (ASYNCcnt) -> s_wait_asynccnt 0.
// ---------------------------------------------------------------------------

// 4) height attention: block = (b, w) column. h_out[b,c,i,w] = sum_j v[c,j]*A[j,i]
__global__ __launch_bounds__(256) void k_attn_h(
    const float* __restrict__ qT, const float* __restrict__ kT,
    const unsigned short* __restrict__ vT_, float* __restrict__ hout) {
  const int b = blockIdx.x >> 8, w = blockIdx.x & 255;
  const int t = threadIdx.x;

  __shared__ __align__(64) float  s_k[NQ * HH];      // 8 KB  [c][i]
  __shared__ __align__(64) bf16_t s_v[CH * HH];      // 32 KB [c][j]
  __shared__ __align__(64) bf16_t s_A[32 * HH];      // 16 KB At[i_local][j]

  // async-stage the contiguous transposed column blocks straight into LDS
  const char* kTb = (const char*)kT + (size_t)(b * WW + w) * (NQ * HH) * 4;
  const char* vTb = (const char*)vT_ + (size_t)(b * WW + w) * (CH * HH) * 2;
#pragma unroll
  for (int it = 0; it < 2; ++it) {
    int q16 = t + it * 256;
    async_copy_b128((char*)s_k + q16 * 16, kTb + q16 * 16);
  }
#pragma unroll
  for (int it = 0; it < 8; ++it) {
    int q16 = t + it * 256;
    async_copy_b128((char*)s_v + q16 * 16, vTb + q16 * 16);
  }
  float qv[NQ];
#pragma unroll
  for (int c = 0; c < NQ; ++c)
    qv[c] = qT[((size_t)(b * WW + w) * NQ + c) * HH + t];
  async_wait0();
  __syncthreads();

  // per-thread j = t: softmax normalizers over i (recompute-based, K=8 cheap)
  float m = -3.0e38f;
  for (int i = 0; i < HH; ++i) {
    float e = 0.f;
#pragma unroll
    for (int c = 0; c < NQ; ++c) e += qv[c] * s_k[c * HH + i];
    m = fmaxf(m, e);
  }
  float ssum = 0.f;
  for (int i = 0; i < HH; ++i) {
    float e = 0.f;
#pragma unroll
    for (int c = 0; c < NQ; ++c) e += qv[c] * s_k[c * HH + i];
    ssum += __expf(e - m);
  }
  const float invs = 1.0f / ssum;

  const int wave = t >> 5, lane = t & 31, hh = lane >> 4, lrow = lane & 15;
  const int cBase = (wave & 3) * 16, iLoc = (wave >> 2) * 16;

  for (int chunk = 0; chunk < 8; ++chunk) {
    const int i0 = chunk * 32;
    for (int il = 0; il < 32; ++il) {
      int i = i0 + il;
      float e = 0.f;
#pragma unroll
      for (int c = 0; c < NQ; ++c) e += qv[c] * s_k[c * HH + i];
      s_A[il * HH + t] = (bf16_t)(__expf(e - m) * invs);
    }
    __syncthreads();

    v8f acc = {};
#pragma unroll
    for (int j0 = 0; j0 < HH; j0 += 32) {
      const bf16_t* vrow = &s_v[(cBase + lrow) * HH + j0 + hh * 16];
      v8bf lo = *(const v8bf*)vrow;
      v8bf hi = *(const v8bf*)(vrow + 8);
      v16bf afrag = __builtin_shufflevector(lo, hi, 0, 1, 2, 3, 4, 5, 6, 7,
                                            8, 9, 10, 11, 12, 13, 14, 15);
      v16bf bfrag = *(const v16bf*)(&s_A[(iLoc + lrow) * HH + j0 + hh * 16]);
      acc = __builtin_amdgcn_wmma_f32_16x16x32_bf16(false, afrag, false, bfrag,
                                                    (short)0, acc, false, false);
    }
    const int iGlob = i0 + iLoc + lrow;
    float* obase = hout + ((size_t)(b * CH + cBase + hh * 8) * HH + iGlob) * WW + w;
#pragma unroll
    for (int r = 0; r < 8; ++r) obase[(size_t)r * PLANE] = acc[r];
    __syncthreads();
  }
}

// 5) width attention + fused epilogue: out = gamma*(h_out+v_out) + yn
__global__ __launch_bounds__(256) void k_attn_w_final(
    const float* __restrict__ qg, const float* __restrict__ kg,
    const unsigned short* __restrict__ vg_, const float* __restrict__ hout,
    const float* __restrict__ yn, const float* __restrict__ gamma,
    float* __restrict__ outp) {
  const int b = blockIdx.x >> 8, hrow = blockIdx.x & 255;
  const int t = threadIdx.x;
  const float g = gamma[0];

  __shared__ __align__(64) float  s_k[NQ * WW];
  __shared__ __align__(64) bf16_t s_v[CH * WW];
  __shared__ __align__(64) bf16_t s_A[32 * WW];

  // async-stage row-contiguous k / v slices into LDS
  const char* kgb = (const char*)kg;
  const char* vgb = (const char*)vg_;
#pragma unroll
  for (int it = 0; it < 2; ++it) {
    int q16 = t + it * 256;
    int c = q16 >> 6, off = (q16 & 63) * 16;
    async_copy_b128((char*)s_k + q16 * 16,
                    kgb + (((size_t)(b * NQ + c) * HH + hrow) * WW) * 4 + off);
  }
#pragma unroll
  for (int it = 0; it < 8; ++it) {
    int q16 = t + it * 256;
    int c = q16 >> 5, off = (q16 & 31) * 16;
    async_copy_b128((char*)s_v + q16 * 16,
                    vgb + (((size_t)(b * CH + c) * HH + hrow) * WW) * 2 + off);
  }
  float qv[NQ];
#pragma unroll
  for (int c = 0; c < NQ; ++c)
    qv[c] = qg[((size_t)(b * NQ + c) * HH + hrow) * WW + t];
  async_wait0();
  __syncthreads();

  float m = -3.0e38f;
  for (int i = 0; i < WW; ++i) {
    float e = 0.f;
#pragma unroll
    for (int c = 0; c < NQ; ++c) e += qv[c] * s_k[c * WW + i];
    m = fmaxf(m, e);
  }
  float ssum = 0.f;
  for (int i = 0; i < WW; ++i) {
    float e = 0.f;
#pragma unroll
    for (int c = 0; c < NQ; ++c) e += qv[c] * s_k[c * WW + i];
    ssum += __expf(e - m);
  }
  const float invs = 1.0f / ssum;

  const int wave = t >> 5, lane = t & 31, hh = lane >> 4, lrow = lane & 15;
  const int cBase = (wave & 3) * 16, iLoc = (wave >> 2) * 16;

  for (int chunk = 0; chunk < 8; ++chunk) {
    const int x0 = chunk * 32;
    for (int il = 0; il < 32; ++il) {
      int i = x0 + il;
      float e = 0.f;
#pragma unroll
      for (int c = 0; c < NQ; ++c) e += qv[c] * s_k[c * WW + i];
      s_A[il * WW + t] = (bf16_t)(__expf(e - m) * invs);
    }
    __syncthreads();

    v8f acc = {};
#pragma unroll
    for (int j0 = 0; j0 < WW; j0 += 32) {
      const bf16_t* vrow = &s_v[(cBase + lrow) * WW + j0 + hh * 16];
      v8bf lo = *(const v8bf*)vrow;
      v8bf hi = *(const v8bf*)(vrow + 8);
      v16bf afrag = __builtin_shufflevector(lo, hi, 0, 1, 2, 3, 4, 5, 6, 7,
                                            8, 9, 10, 11, 12, 13, 14, 15);
      v16bf bfrag = *(const v16bf*)(&s_A[(iLoc + lrow) * WW + j0 + hh * 16]);
      acc = __builtin_amdgcn_wmma_f32_16x16x32_bf16(false, afrag, false, bfrag,
                                                    (short)0, acc, false, false);
    }
    const int xGlob = x0 + iLoc + lrow;
    const size_t base = ((size_t)(b * CH + cBase + hh * 8) * HH + hrow) * WW + xGlob;
#pragma unroll
    for (int r = 0; r < 8; ++r) {
      size_t idx = base + (size_t)r * PLANE;
      outp[idx] = g * (hout[idx] + acc[r]) + yn[idx];
    }
    __syncthreads();
  }
}

// 6) 2x2 maxpool
__global__ void k_pool(const float* __restrict__ outp, float* __restrict__ down) {
  int idx = blockIdx.x * 256 + threadIdx.x;
  if (idx >= NB * CH * 128 * 128) return;
  int xo = idx & 127;
  int rest = idx >> 7;
  int yo = rest & 127;
  int bc = rest >> 7;
  const float* src = outp + ((size_t)bc * HH + yo * 2) * WW + xo * 2;
  down[idx] = fmaxf(fmaxf(src[0], src[1]), fmaxf(src[WW], src[WW + 1]));
}

// ---------------------------------------------------------------------------
extern "C" void kernel_launch(void* const* d_in, const int* in_sizes, int n_in,
                              void* d_out, int out_size, void* d_ws, size_t ws_size,
                              hipStream_t stream) {
  const float* x       = (const float*)d_in[0];
  const float* hconv_w = (const float*)d_in[1];
  const float* hconv_b = (const float*)d_in[2];
  const float* vconv_w = (const float*)d_in[3];
  const float* vconv_b = (const float*)d_in[4];
  const float* pw_w    = (const float*)d_in[5];
  const float* pw_b    = (const float*)d_in[6];
  const float* bn_w    = (const float*)d_in[7];
  const float* bn_b    = (const float*)d_in[8];
  const float* q_w     = (const float*)d_in[9];
  const float* q_b     = (const float*)d_in[10];
  const float* k_w     = (const float*)d_in[11];
  const float* k_b     = (const float*)d_in[12];
  const float* v_w     = (const float*)d_in[13];
  const float* v_b     = (const float*)d_in[14];
  const float* gamma   = (const float*)d_in[15];

  char* ws = (char*)d_ws;
  const size_t SZ_Y = (size_t)NB * CH * PLANE * 4;  // 67,108,864
  const size_t SZ_Q = (size_t)NB * NQ * PLANE * 4;  //  8,388,608
  const size_t SZ_V = (size_t)NB * CH * PLANE * 2;  // 33,554,432
  size_t off = 0;
  float* stats = (float*)(ws + off); off += 4096;          // 2KB used + pad
  float* coefA = (float*)(ws + off); off += 1024;
  float* coefB = (float*)(ws + off); off += 1024;
  off = (off + 255) & ~(size_t)255;
  float* ybuf  = (float*)(ws + off); off += SZ_Y;
  float* qbuf  = (float*)(ws + off); off += SZ_Q;
  float* kbuf  = (float*)(ws + off); off += SZ_Q;
  unsigned short* vbuf = (unsigned short*)(ws + off); off += SZ_V;
  float* hbuf  = (float*)(ws + off); off += SZ_Y;
  float* qTbuf = (float*)(ws + off); off += SZ_Q;
  float* kTbuf = (float*)(ws + off); off += SZ_Q;
  unsigned short* vTbuf = (unsigned short*)(ws + off); off += SZ_V;

  float* outp = (float*)d_out;
  float* down = outp + (size_t)NB * CH * PLANE;

  k_zero<<<2, 256, 0, stream>>>(stats);
  k_conv<<<NB * HH, 256, 0, stream>>>(x, hconv_w, hconv_b, vconv_w, vconv_b,
                                      pw_w, pw_b, ybuf, stats);
  k_coef<<<1, 64, 0, stream>>>(stats, bn_w, bn_b, coefA, coefB);
  k_norm_proj<<<NB * HH, 256, 0, stream>>>(ybuf, q_w, q_b, k_w, k_b, v_w, v_b,
                                           coefA, coefB, qbuf, kbuf, vbuf,
                                           qTbuf, kTbuf, vTbuf);
  k_attn_h<<<NB * WW, 256, 0, stream>>>(qTbuf, kTbuf, vTbuf, hbuf);
  k_attn_w_final<<<NB * HH, 256, 0, stream>>>(qbuf, kbuf, vbuf, hbuf, ybuf,
                                              gamma, outp);
  k_pool<<<(NB * CH * 128 * 128) / 256, 256, 0, stream>>>(outp, down);
}
